// DetectiveNetwork_72834055406050
// MI455X (gfx1250) — matrix operand order repaired and checked by reading, not compiled, not run on hardware
//
#include <hip/hip_runtime.h>
#include <hip/hip_bf16.h>

// ---------------------------------------------------------------------------
// MI455X (gfx1250) implementation: every GEMM stage lowered to
// v_wmma_f32_16x16x32_bf16 (wave32 WMMA, f32 accumulate); bf16-sourced LDS
// staging uses CDNA5 global_load_async_to_lds_b128 (ASYNCcnt-tracked).
// ---------------------------------------------------------------------------

typedef __attribute__((__ext_vector_type__(16))) __bf16 v16bf;
typedef __attribute__((__ext_vector_type__(8)))  float  v8f;

#define NROW   65536   // T*B = 256*256
#define LDH    40      // LDS row stride (halfwords); 80B rows keep B128 dests 16B-aligned

__device__ __forceinline__ unsigned short f2bf(float x) {
    union { __bf16 b; unsigned short s; } u;
    u.b = (__bf16)x;                    // native RNE convert on gfx1250
    return u.s;
}

__device__ __forceinline__ void stage16_f32(unsigned short* d, const float* s, float scale) {
    const float4* p = (const float4*)s;
#pragma unroll
    for (int q = 0; q < 4; ++q) {
        float4 v = p[q];
        d[4*q+0] = f2bf(v.x * scale);
        d[4*q+1] = f2bf(v.y * scale);
        d[4*q+2] = f2bf(v.z * scale);
        d[4*q+3] = f2bf(v.w * scale);
    }
}

// CDNA5 async global->LDS copy: 32 bytes per lane (2 x B128), ASYNCcnt-tracked.
// ldsoff: per-lane LDS byte offset (16B aligned); goff: per-lane global byte
// offset from uniform base (GVS addressing mode).
__device__ __forceinline__ void async_copy32(const void* base, unsigned goff, unsigned ldsoff) {
    asm volatile(
        "global_load_async_to_lds_b128 %0, %1, %2\n\t"
        "global_load_async_to_lds_b128 %0, %1, %2 offset:16"
        :: "v"(ldsoff), "v"(goff), "s"((unsigned long long)(size_t)base)
        : "memory");
}
__device__ __forceinline__ void async_wait0() {
    asm volatile("s_wait_asynccnt 0x0" ::: "memory");
}

// A-matrix fragment, 16x32 bf16 (ISA 7.12.2): lane<16 -> M=lane, K in {0..7,16..23};
// lane>=16 -> M=lane-16, K in {8..15,24..31}. LDS layout ls[row][k], row stride ld.
__device__ __forceinline__ v16bf frag_ld_a(const unsigned short* base, int ld, int lane) {
    const unsigned short* p = base + (lane & 15) * ld;
    const int ko = (lane & 16) ? 8 : 0;
    union { v16bf v; unsigned u[8]; } r;
#pragma unroll
    for (int g = 0; g < 4; ++g) {
        r.u[g]     = *(const unsigned*)(p + ko + 2*g);
        r.u[4 + g] = *(const unsigned*)(p + 16 + ko + 2*g);
    }
    return r.v;
}

// B-matrix fragment, 32x16 bf16: lane<16 -> N=lane, K in {0..7,16..23}; lane>=16 -> N=lane-16, K in {8..15,24..31}.
// LDS layout ls[n][k], row stride ld.
__device__ __forceinline__ v16bf frag_ld_b(const unsigned short* base, int ld, int lane) {
    const unsigned short* p = base + (lane & 15) * ld;
    const int ko = (lane & 16) ? 8 : 0;
    union { v16bf v; unsigned u[8]; } r;
#pragma unroll
    for (int g = 0; g < 8; ++g)
        r.u[g] = *(const unsigned*)(p + ko + 2*g + ((g >= 4) ? 8 : 0));
    return r.v;
}

__device__ __forceinline__ v8f wmma_bf16(v16bf a, v16bf b, v8f c) {
    return __builtin_amdgcn_wmma_f32_16x16x32_bf16(false, a, false, b, (short)0, c, false, false);
}

__device__ __forceinline__ float wred_sum(float v) {
#pragma unroll
    for (int m = 16; m >= 1; m >>= 1) v += __shfl_xor(v, m, 32);
    return v;
}
__device__ __forceinline__ float wred_max(float v) {
#pragma unroll
    for (int m = 16; m >= 1; m >>= 1) v = fmaxf(v, __shfl_xor(v, m, 32));
    return v;
}

// ---------------------------------------------------------------------------
// Generic activation x weight^T GEMM.  C[M,N] = A[M,K] @ W[N,K]^T (+epilogue)
// Tile 128x128, 8 waves, each wave 32x64 (2x4 WMMA tiles), K chunk 32.
// AMODE: 0=f32 A, 1=U_s masked by seq length, 2=bf16 A (async-to-LDS),
//        3=concat(mask*U_s, A2)
// EMODE: 0=+bias, 1=+bias+PE(batch-dim quirk), 2=+bias+Res, 3=relu->bf16, 4=relu->f32
// ---------------------------------------------------------------------------
struct GemmArgs {
    const float* A;
    const unsigned short* Abf;
    const float* A2;
    const float* W;
    const float* bias;
    const float* Res;
    const int*   L;
    float* Cf;
    unsigned short* Cbf;
    int N, K, lda;
};

template <int AMODE, int EMODE>
__global__ __launch_bounds__(256) void k_gemm(GemmArgs a) {
    __shared__ unsigned short lsA[128][LDH];
    __shared__ unsigned short lsB[128][LDH];
    const int m0 = blockIdx.y * 128, n0 = blockIdx.x * 128;
    const int tid = threadIdx.x, lane = tid & 31, wave = tid >> 5;
    const int wr = wave & 3, wc = wave >> 2;

    const v8f vz = {0.f,0.f,0.f,0.f,0.f,0.f,0.f,0.f};
    v8f acc[2][4];
#pragma unroll
    for (int i = 0; i < 2; ++i)
#pragma unroll
        for (int j = 0; j < 4; ++j) acc[i][j] = vz;

    const int row = tid >> 1, half = tid & 1;
    const int gm = m0 + row;
    float amask = 1.f;
    if (AMODE == 1 || AMODE == 3) {
        int t = gm >> 8, bb = gm & 255;
        amask = (t < a.L[bb]) ? 1.f : 0.f;
    }

    for (int kb = 0; kb < a.K; kb += 32) {
        const int kg = kb + half * 16;
        unsigned short* da = &lsA[row][half * 16];
        if (AMODE == 0) {
            stage16_f32(da, a.A + (size_t)gm * a.lda + kg, 1.f);
        } else if (AMODE == 1) {
            stage16_f32(da, a.A + (size_t)gm * a.lda + kg, amask);
        } else if (AMODE == 2) {
            // async DMA: 32B of bf16 straight into LDS, no VGPR round trip
            const unsigned goff = (unsigned)(((size_t)gm * a.lda + kg) * 2);
            async_copy32(a.Abf, goff, (unsigned)(size_t)da);
        } else {
            if (kg < 256) stage16_f32(da, a.A  + (size_t)gm * 256 + kg, amask);
            else          stage16_f32(da, a.A2 + (size_t)gm * 256 + (kg - 256), 1.f);
        }
        stage16_f32(&lsB[row][half * 16], a.W + (size_t)(n0 + row) * a.K + kg, 1.f);
        if (AMODE == 2) async_wait0();
        __syncthreads();
        if (kb + 32 < a.K) {
            __builtin_prefetch(a.W + (size_t)(n0 + row) * a.K + kg + 32, 0, 0);
        }
        v16bf af[2], bf[4];
#pragma unroll
        for (int i = 0; i < 2; ++i) af[i] = frag_ld_a(&lsA[wr * 32 + i * 16][0], LDH, lane);
#pragma unroll
        for (int j = 0; j < 4; ++j) bf[j] = frag_ld_b(&lsB[wc * 64 + j * 16][0], LDH, lane);
#pragma unroll
        for (int i = 0; i < 2; ++i)
#pragma unroll
            for (int j = 0; j < 4; ++j) acc[i][j] = wmma_bf16(af[i], bf[j], acc[i][j]);
        __syncthreads();
    }

    const float kPE = -9.210340371976184f / 256.0f;   // -ln(10000)/F
#pragma unroll
    for (int i = 0; i < 2; ++i)
#pragma unroll
        for (int j = 0; j < 4; ++j) {
            union { v8f v; float f[8]; } u; u.v = acc[i][j];
            const int gn = n0 + wc * 64 + j * 16 + (lane & 15);
            const int mb = m0 + wr * 32 + i * 16 + ((lane >> 4) << 3);
            const float bs = a.bias[gn];
#pragma unroll
            for (int g = 0; g < 8; ++g) {
                const int gmm = mb + g;
                float v = u.f[g] + bs;
                if (EMODE == 1) {
                    const int bcol = gmm & 255;           // PE indexed by batch (quirk)
                    const float dv = __expf((float)(gn & ~1) * kPE);
                    const float ang = (float)bcol * dv;
                    v += (gn & 1) ? __cosf(ang) : __sinf(ang);
                }
                if (EMODE == 2) v += a.Res[(size_t)gmm * a.N + gn];
                if (EMODE == 3) {
                    a.Cbf[(size_t)gmm * a.N + gn] = f2bf(fmaxf(v, 0.f));
                } else if (EMODE == 4) {
                    a.Cf[(size_t)gmm * a.N + gn] = fmaxf(v, 0.f);
                } else {
                    a.Cf[(size_t)gmm * a.N + gn] = v;
                }
            }
        }
}

// ---------------------------------------------------------------------------
// Attention scores: S[z, s, t] = (Q_z[s,:] . K_z[t,:]) / 8,  z = b*4 + h, K=64.
// ---------------------------------------------------------------------------
__global__ __launch_bounds__(256) void k_attn_scores(const float* __restrict__ qkv,
                                                     float* __restrict__ S) {
    __shared__ unsigned short lsA[128][LDH];
    __shared__ unsigned short lsB[128][LDH];
    const int z = blockIdx.z, b = z >> 2, h = z & 3;
    const float* qb = qkv + (size_t)b * 768 + h * 64;
    const float* kbp = qb + 256;
    const int m0 = blockIdx.y * 128, n0 = blockIdx.x * 128;
    const int tid = threadIdx.x, lane = tid & 31, wave = tid >> 5;
    const int wr = wave & 3, wc = wave >> 2;
    const v8f vz = {0.f,0.f,0.f,0.f,0.f,0.f,0.f,0.f};
    v8f acc[2][4];
#pragma unroll
    for (int i = 0; i < 2; ++i)
#pragma unroll
        for (int j = 0; j < 4; ++j) acc[i][j] = vz;
    const int row = tid >> 1, half = tid & 1;
    for (int kb = 0; kb < 64; kb += 32) {
        const int kg = kb + half * 16;
        stage16_f32(&lsA[row][half * 16], qb  + (size_t)(m0 + row) * 196608 + kg, 1.f);
        stage16_f32(&lsB[row][half * 16], kbp + (size_t)(n0 + row) * 196608 + kg, 1.f);
        __syncthreads();
        v16bf af[2], bf[4];
#pragma unroll
        for (int i = 0; i < 2; ++i) af[i] = frag_ld_a(&lsA[wr * 32 + i * 16][0], LDH, lane);
#pragma unroll
        for (int j = 0; j < 4; ++j) bf[j] = frag_ld_b(&lsB[wc * 64 + j * 16][0], LDH, lane);
#pragma unroll
        for (int i = 0; i < 2; ++i)
#pragma unroll
            for (int j = 0; j < 4; ++j) acc[i][j] = wmma_bf16(af[i], bf[j], acc[i][j]);
        __syncthreads();
    }
#pragma unroll
    for (int i = 0; i < 2; ++i)
#pragma unroll
        for (int j = 0; j < 4; ++j) {
            union { v8f v; float f[8]; } u; u.v = acc[i][j];
            const int gn = n0 + wc * 64 + j * 16 + (lane & 15);
            const int mb = m0 + wr * 32 + i * 16 + ((lane >> 4) << 3);
#pragma unroll
            for (int g = 0; g < 8; ++g)
                S[(size_t)z * 65536 + (size_t)(mb + g) * 256 + gn] = u.f[g] * 0.125f;
        }
}

// ---------------------------------------------------------------------------
// Attention output: O[z, s, d] = sum_t P[z,s,t] * V_z[t,d],  N=64, K=256.
// P (bf16 probabilities) staged via async-to-LDS DMA.
// ---------------------------------------------------------------------------
__global__ __launch_bounds__(256) void k_attn_o(const unsigned short* __restrict__ P,
                                                const float* __restrict__ qkv,
                                                float* __restrict__ AO) {
    __shared__ unsigned short lsA[128][LDH];
    __shared__ unsigned short lsB[64][LDH];
    const int z = blockIdx.z, b = z >> 2, h = z & 3;
    const int m0 = blockIdx.y * 128;
    const int tid = threadIdx.x, lane = tid & 31, wave = tid >> 5;
    const int wr = wave & 3, wc = wave >> 2;
    const v8f vz = {0.f,0.f,0.f,0.f,0.f,0.f,0.f,0.f};
    v8f acc[2][2];
#pragma unroll
    for (int i = 0; i < 2; ++i)
#pragma unroll
        for (int j = 0; j < 2; ++j) acc[i][j] = vz;

    for (int kb = 0; kb < 256; kb += 32) {
        { // A: bf16 probabilities -> async DMA into LDS
            const int row = tid >> 1, half = tid & 1;
            const unsigned goff =
                (unsigned)((((size_t)z * 256 + m0 + row) * 256 + kb + half * 16) * 2);
            async_copy32(P, goff, (unsigned)(size_t)&lsA[row][half * 16]);
        }
        { // B: V rows (k=t, n=d) -> transposed stage lsB[n][k]
            const int kk = tid >> 3, nb = (tid & 7) * 8;
            const float* src = qkv + ((size_t)(kb + kk) * 256 + b) * 768 + 512 + h * 64 + nb;
            float4 v0 = *(const float4*)src;
            float4 v1 = *(const float4*)(src + 4);
            lsB[nb + 0][kk] = f2bf(v0.x); lsB[nb + 1][kk] = f2bf(v0.y);
            lsB[nb + 2][kk] = f2bf(v0.z); lsB[nb + 3][kk] = f2bf(v0.w);
            lsB[nb + 4][kk] = f2bf(v1.x); lsB[nb + 5][kk] = f2bf(v1.y);
            lsB[nb + 6][kk] = f2bf(v1.z); lsB[nb + 7][kk] = f2bf(v1.w);
        }
        async_wait0();
        __syncthreads();
        v16bf af[2], bf[2];
#pragma unroll
        for (int i = 0; i < 2; ++i) af[i] = frag_ld_a(&lsA[wr * 32 + i * 16][0], LDH, lane);
#pragma unroll
        for (int j = 0; j < 2; ++j) bf[j] = frag_ld_b(&lsB[wc * 32 + j * 16][0], LDH, lane);
#pragma unroll
        for (int i = 0; i < 2; ++i)
#pragma unroll
            for (int j = 0; j < 2; ++j) acc[i][j] = wmma_bf16(af[i], bf[j], acc[i][j]);
        __syncthreads();
    }
#pragma unroll
    for (int i = 0; i < 2; ++i)
#pragma unroll
        for (int j = 0; j < 2; ++j) {
            union { v8f v; float f[8]; } u; u.v = acc[i][j];
            const int gn = wc * 32 + j * 16 + (lane & 15);         // d in [0,64)
            const int mb = m0 + wr * 32 + i * 16 + ((lane >> 4) << 3);
#pragma unroll
            for (int g = 0; g < 8; ++g)
                AO[((size_t)(mb + g) * 256 + b) * 256 + h * 64 + gn] = u.f[g];
        }
}

// ---------------------------------------------------------------------------
// Pooling scores: E[b, t, i] = q[t,b,:] . U_p[i,b,:]  (masked i>=L[b] -> -1e30)
// ---------------------------------------------------------------------------
__global__ __launch_bounds__(256) void k_pool_scores(const float* __restrict__ q,
                                                     const float* __restrict__ U_p,
                                                     const int* __restrict__ L,
                                                     float* __restrict__ S) {
    __shared__ unsigned short lsA[128][LDH];
    __shared__ unsigned short lsB[128][LDH];
    const int b = blockIdx.z;
    const int m0 = blockIdx.y * 128, n0 = blockIdx.x * 128;
    const int tid = threadIdx.x, lane = tid & 31, wave = tid >> 5;
    const int wr = wave & 3, wc = wave >> 2;
    const v8f vz = {0.f,0.f,0.f,0.f,0.f,0.f,0.f,0.f};
    v8f acc[2][4];
#pragma unroll
    for (int i = 0; i < 2; ++i)
#pragma unroll
        for (int j = 0; j < 4; ++j) acc[i][j] = vz;
    const int row = tid >> 1, half = tid & 1;
    for (int kb = 0; kb < 256; kb += 32) {
        const int kg = kb + half * 16;
        stage16_f32(&lsA[row][half * 16], q   + (size_t)(m0 + row) * 65536 + (size_t)b * 256 + kg, 1.f);
        stage16_f32(&lsB[row][half * 16], U_p + (size_t)(n0 + row) * 65536 + (size_t)b * 256 + kg, 1.f);
        __syncthreads();
        v16bf af[2], bf[4];
#pragma unroll
        for (int i = 0; i < 2; ++i) af[i] = frag_ld_a(&lsA[wr * 32 + i * 16][0], LDH, lane);
#pragma unroll
        for (int j = 0; j < 4; ++j) bf[j] = frag_ld_b(&lsB[wc * 64 + j * 16][0], LDH, lane);
#pragma unroll
        for (int i = 0; i < 2; ++i)
#pragma unroll
            for (int j = 0; j < 4; ++j) acc[i][j] = wmma_bf16(af[i], bf[j], acc[i][j]);
        __syncthreads();
    }
    const int Lb = L[b];
#pragma unroll
    for (int i = 0; i < 2; ++i)
#pragma unroll
        for (int j = 0; j < 4; ++j) {
            union { v8f v; float f[8]; } u; u.v = acc[i][j];
            const int gn = n0 + wc * 64 + j * 16 + (lane & 15);    // i (turn)
            const int mb = m0 + wr * 32 + i * 16 + ((lane >> 4) << 3);
#pragma unroll
            for (int g = 0; g < 8; ++g) {
                float v = (gn >= Lb) ? -1e30f : u.f[g];
                S[((size_t)b * 256 + (mb + g)) * 256 + gn] = v;
            }
        }
}

// ---------------------------------------------------------------------------
// Pooling gather: R[t,b,f] = sum_i A[b,t,i] * (U_p[i,b,f] * mask[i,b])
// A (bf16 probabilities) staged via async-to-LDS DMA.
// ---------------------------------------------------------------------------
__global__ __launch_bounds__(256) void k_pool_r(const unsigned short* __restrict__ P,
                                                const float* __restrict__ U_p,
                                                const int* __restrict__ L,
                                                float* __restrict__ R) {
    __shared__ unsigned short lsA[128][LDH];
    __shared__ unsigned short lsB[128][LDH];
    const int b = blockIdx.z;
    const int m0 = blockIdx.y * 128, n0 = blockIdx.x * 128;
    const int tid = threadIdx.x, lane = tid & 31, wave = tid >> 5;
    const int wr = wave & 3, wc = wave >> 2;
    const int Lb = L[b];
    const v8f vz = {0.f,0.f,0.f,0.f,0.f,0.f,0.f,0.f};
    v8f acc[2][4];
#pragma unroll
    for (int i = 0; i < 2; ++i)
#pragma unroll
        for (int j = 0; j < 4; ++j) acc[i][j] = vz;

    for (int kb = 0; kb < 256; kb += 32) {
        { // A: bf16 probabilities, rows t, cols i -> async DMA into LDS
            const int row = tid >> 1, half = tid & 1;
            const unsigned goff =
                (unsigned)((((size_t)b * 256 + m0 + row) * 256 + kb + half * 16) * 2);
            async_copy32(P, goff, (unsigned)(size_t)&lsA[row][half * 16]);
        }
        { // B: U_p rows i (=k), cols f (=n); masked; transposed stage lsB[n][k]
            const int kk = tid >> 3, nb = (tid & 7) * 16;
            const int ii = kb + kk;
            const float msk = (ii < Lb) ? 1.f : 0.f;
            const float4* src = (const float4*)(U_p + ((size_t)ii * 256 + b) * 256 + n0 + nb);
#pragma unroll
            for (int q = 0; q < 4; ++q) {
                float4 v = src[q];
                lsB[nb + 4*q + 0][kk] = f2bf(v.x * msk);
                lsB[nb + 4*q + 1][kk] = f2bf(v.y * msk);
                lsB[nb + 4*q + 2][kk] = f2bf(v.z * msk);
                lsB[nb + 4*q + 3][kk] = f2bf(v.w * msk);
            }
        }
        async_wait0();
        __syncthreads();
        v16bf af[2], bf[4];
#pragma unroll
        for (int i = 0; i < 2; ++i) af[i] = frag_ld_a(&lsA[wr * 32 + i * 16][0], LDH, lane);
#pragma unroll
        for (int j = 0; j < 4; ++j) bf[j] = frag_ld_b(&lsB[wc * 64 + j * 16][0], LDH, lane);
#pragma unroll
        for (int i = 0; i < 2; ++i)
#pragma unroll
            for (int j = 0; j < 4; ++j) acc[i][j] = wmma_bf16(af[i], bf[j], acc[i][j]);
        __syncthreads();
    }
#pragma unroll
    for (int i = 0; i < 2; ++i)
#pragma unroll
        for (int j = 0; j < 4; ++j) {
            union { v8f v; float f[8]; } u; u.v = acc[i][j];
            const int gn = n0 + wc * 64 + j * 16 + (lane & 15);    // f
            const int mb = m0 + wr * 32 + i * 16 + ((lane >> 4) << 3);
#pragma unroll
            for (int g = 0; g < 8; ++g)
                R[((size_t)(mb + g) * 256 + b) * 256 + gn] = u.f[g];
        }
}

// ---------------------------------------------------------------------------
// Row softmax over 256 cols, one wave per row, output bf16 probabilities.
// ---------------------------------------------------------------------------
__global__ __launch_bounds__(256) void k_softmax(const float* __restrict__ S,
                                                 unsigned short* __restrict__ P,
                                                 int nrows) {
    const int wave = threadIdx.x >> 5, lane = threadIdx.x & 31;
    const int row = blockIdx.x * 8 + wave;
    if (row >= nrows) return;
    const float* p = S + (size_t)row * 256;
    float v[8];
    float m = -3.4e38f;
#pragma unroll
    for (int i = 0; i < 8; ++i) { v[i] = p[lane + i * 32]; m = fmaxf(m, v[i]); }
    m = wred_max(m);
    float s = 0.f;
#pragma unroll
    for (int i = 0; i < 8; ++i) { v[i] = __expf(v[i] - m); s += v[i]; }
    s = wred_sum(s);
    const float inv = 1.f / s;
    unsigned short* q = P + (size_t)row * 256;
#pragma unroll
    for (int i = 0; i < 8; ++i) q[lane + i * 32] = f2bf(v[i] * inv);
}

// ---------------------------------------------------------------------------
// LayerNorm over 256 cols; one wave per row.  ADDH: Y = H + LN(X) (q = h + h2)
// ---------------------------------------------------------------------------
template <int ADDH>
__global__ __launch_bounds__(256) void k_ln(const float* __restrict__ X,
                                            const float* __restrict__ gw,
                                            const float* __restrict__ bw,
                                            const float* __restrict__ H,
                                            float* __restrict__ Y) {
    const int wave = threadIdx.x >> 5, lane = threadIdx.x & 31;
    const int row = blockIdx.x * 8 + wave;
    const float* p = X + (size_t)row * 256;
    float v[8];
    float s = 0.f, sq = 0.f;
#pragma unroll
    for (int i = 0; i < 8; ++i) { v[i] = p[lane + i * 32]; s += v[i]; sq += v[i] * v[i]; }
    s = wred_sum(s); sq = wred_sum(sq);
    const float mean = s * (1.f / 256.f);
    const float var = sq * (1.f / 256.f) - mean * mean;
    const float inv = rsqrtf(var + 1e-5f);
    float* q = Y + (size_t)row * 256;
#pragma unroll
    for (int i = 0; i < 8; ++i) {
        const int c = lane + i * 32;
        float y = (v[i] - mean) * inv * gw[c] + bw[c];
        if (ADDH) y += H[(size_t)row * 256 + c];
        q[c] = y;
    }
}

// ---------------------------------------------------------------------------
// Classifier head: logits (K=512, N=7) -> log_softmax -> ragged scatter.
// ---------------------------------------------------------------------------
__global__ __launch_bounds__(256) void k_head(const float* __restrict__ hidden,
                                              const float* __restrict__ Wsm,
                                              const float* __restrict__ bsm,
                                              const int* __restrict__ L,
                                              const int* __restrict__ pre,
                                              float* __restrict__ out) {
    __shared__ float w[7 * 512];
    const int tid = threadIdx.x;
    for (int i = tid; i < 7 * 512; i += 256) w[i] = Wsm[i];
    __syncthreads();
    const int row = blockIdx.x * 256 + tid;
    const int t = row >> 8, b = row & 255;
    if (t >= L[b]) return;
    const float* hp = hidden + (size_t)row * 512;
    float lg[7];
#pragma unroll
    for (int c = 0; c < 7; ++c) lg[c] = bsm[c];
    for (int k = 0; k < 512; k += 4) {
        float4 hv = *(const float4*)(hp + k);
#pragma unroll
        for (int c = 0; c < 7; ++c) {
            const float* wc = w + c * 512 + k;
            lg[c] += hv.x * wc[0] + hv.y * wc[1] + hv.z * wc[2] + hv.w * wc[3];
        }
    }
    float m = lg[0];
#pragma unroll
    for (int c = 1; c < 7; ++c) m = fmaxf(m, lg[c]);
    float s = 0.f;
#pragma unroll
    for (int c = 0; c < 7; ++c) s += __expf(lg[c] - m);
    const float lse = m + __logf(s);
    float* o = out + (size_t)(pre[b] + t) * 7;
#pragma unroll
    for (int c = 0; c < 7; ++c) o[c] = lg[c] - lse;
}

__global__ void k_prefix(const int* __restrict__ L, int* __restrict__ pre) {
    __shared__ int s[256];
    const int tid = threadIdx.x;
    s[tid] = L[tid];
    __syncthreads();
    for (int off = 1; off < 256; off <<= 1) {
        int v = (tid >= off) ? s[tid - off] : 0;
        __syncthreads();
        s[tid] += v;
        __syncthreads();
    }
    pre[tid] = s[tid] - L[tid];
}

// ---------------------------------------------------------------------------
extern "C" void kernel_launch(void* const* d_in, const int* in_sizes, int n_in,
                              void* d_out, int out_size, void* d_ws, size_t ws_size,
                              hipStream_t stream) {
    (void)in_sizes; (void)n_in; (void)out_size; (void)ws_size;
    const float* U_s   = (const float*)d_in[0];
    const float* U_p   = (const float*)d_in[1];
    const float* W_emb = (const float*)d_in[2];
    const float* b_emb = (const float*)d_in[3];
    const float* Wqkv  = (const float*)d_in[4];
    const float* bqkv  = (const float*)d_in[5];
    const float* Wo    = (const float*)d_in[6];
    const float* bo    = (const float*)d_in[7];
    const float* ln1g  = (const float*)d_in[8];
    const float* ln1b  = (const float*)d_in[9];
    const float* W1    = (const float*)d_in[10];
    const float* b1    = (const float*)d_in[11];
    const float* W2    = (const float*)d_in[12];
    const float* b2    = (const float*)d_in[13];
    const float* ln2g  = (const float*)d_in[14];
    const float* ln2b  = (const float*)d_in[15];
    const float* W_fc  = (const float*)d_in[16];
    const float* b_fc  = (const float*)d_in[17];
    const float* W_sm  = (const float*)d_in[18];
    const float* b_sm  = (const float*)d_in[19];
    const int*   L     = (const int*)d_in[20];

    char* w = (char*)d_ws;
    size_t off = 0;
    auto take = [&](size_t n) -> void* {
        void* p = w + off;
        off += (n + 255) & ~(size_t)255;
        return p;
    };
    float* b_h      = (float*)take((size_t)NROW * 256 * 4);
    float* b_qkv    = (float*)take((size_t)NROW * 768 * 4);
    float* b_hidden = b_qkv;                       // alias: qkv dead before fc
    float* b_sum1   = (float*)take((size_t)NROW * 256 * 4);
    float* b_h1     = (float*)take((size_t)NROW * 256 * 4);
    unsigned short* b_ffmid = (unsigned short*)take((size_t)NROW * 2048 * 2);
    float* b_sum2   = (float*)take((size_t)NROW * 256 * 4);   // h1+ff, then q
    float* b_scores = (float*)take((size_t)1024 * 65536 * 4);
    unsigned short* b_probs = (unsigned short*)take((size_t)1024 * 65536 * 2);
    float* b_ao     = (float*)take((size_t)NROW * 256 * 4);   // attn out, then r
    int*   b_prefix = (int*)take(256 * 4);

    k_prefix<<<dim3(1), dim3(256), 0, stream>>>(L, b_prefix);

    GemmArgs g{};
    // 1. h = mask*U_s @ W_emb^T + b_emb + PE(batch)
    g = GemmArgs{}; g.A = U_s; g.W = W_emb; g.bias = b_emb; g.L = L;
    g.Cf = b_h; g.N = 256; g.K = 256; g.lda = 256;
    k_gemm<1, 1><<<dim3(2, 512), dim3(256), 0, stream>>>(g);

    // 2. qkv = h @ Wqkv^T + bqkv
    g = GemmArgs{}; g.A = b_h; g.W = Wqkv; g.bias = bqkv;
    g.Cf = b_qkv; g.N = 768; g.K = 256; g.lda = 256;
    k_gemm<0, 0><<<dim3(6, 512), dim3(256), 0, stream>>>(g);

    // 3-5. attention
    k_attn_scores<<<dim3(2, 2, 1024), dim3(256), 0, stream>>>(b_qkv, b_scores);
    k_softmax<<<dim3(32768), dim3(256), 0, stream>>>(b_scores, b_probs, 262144);
    k_attn_o<<<dim3(1, 2, 1024), dim3(256), 0, stream>>>(b_probs, b_qkv, b_ao);

    // 6. sum1 = h + (attn_o @ Wo^T + bo)
    g = GemmArgs{}; g.A = b_ao; g.W = Wo; g.bias = bo; g.Res = b_h;
    g.Cf = b_sum1; g.N = 256; g.K = 256; g.lda = 256;
    k_gemm<0, 2><<<dim3(2, 512), dim3(256), 0, stream>>>(g);

    // 7. h1 = LN(sum1)
    k_ln<0><<<dim3(8192), dim3(256), 0, stream>>>(b_sum1, ln1g, ln1b, nullptr, b_h1);

    // 8. ffmid = relu(h1 @ W1^T + b1)  (bf16)
    g = GemmArgs{}; g.A = b_h1; g.W = W1; g.bias = b1;
    g.Cbf = b_ffmid; g.N = 2048; g.K = 256; g.lda = 256;
    k_gemm<0, 3><<<dim3(16, 512), dim3(256), 0, stream>>>(g);

    // 9. sum2 = h1 + (ffmid @ W2^T + b2)   (bf16 A staged via async DMA)
    g = GemmArgs{}; g.Abf = b_ffmid; g.W = W2; g.bias = b2; g.Res = b_h1;
    g.Cf = b_sum2; g.N = 256; g.K = 2048; g.lda = 2048;
    k_gemm<2, 2><<<dim3(2, 512), dim3(256), 0, stream>>>(g);

    // 10. q = h + LN(sum2)   (in place in b_sum2)
    k_ln<1><<<dim3(8192), dim3(256), 0, stream>>>(b_sum2, ln2g, ln2b, b_h, b_sum2);

    // 11-13. attention pooling over turns
    k_pool_scores<<<dim3(2, 2, 256), dim3(256), 0, stream>>>(b_sum2, U_p, L, b_scores);
    k_softmax<<<dim3(8192), dim3(256), 0, stream>>>(b_scores, b_probs, 65536);
    k_pool_r<<<dim3(2, 2, 256), dim3(256), 0, stream>>>(b_probs, U_p, L, b_ao);

    // 14. hidden = relu(concat(mask*U_s, r) @ W_fc^T + b_fc)
    g = GemmArgs{}; g.A = U_s; g.A2 = b_ao; g.W = W_fc; g.bias = b_fc; g.L = L;
    g.Cf = b_hidden; g.N = 512; g.K = 512; g.lda = 256;
    k_gemm<3, 4><<<dim3(4, 512), dim3(256), 0, stream>>>(g);

    // 15. head + log_softmax + ragged scatter
    k_head<<<dim3(256), dim3(256), 0, stream>>>(b_hidden, W_sm, b_sm, L, b_prefix,
                                                (float*)d_out);
}
